// SVDLUT_36498632082027
// MI455X (gfx1250) — matrix operand order repaired and checked by reading, not compiled, not run on hardware
//
#include <hip/hip_runtime.h>
#include <hip/hip_bf16.h>

// ---------------- constants from the reference ----------------
#define V_ 17
#define S_ 8
#define NSVD 280            // V*S + S + S*V
#define CPG_ 2
#define B_ 2
#define IMH 1080
#define IMW 1920

typedef __attribute__((ext_vector_type(16))) _Float16 v16h;
typedef __attribute__((ext_vector_type(8)))  float    v8f;

__device__ __forceinline__ float lrelu_(float t) { return t >= 0.f ? t : 0.2f * t; }
__device__ __forceinline__ float clamp01(float v) { return fminf(fmaxf(v, 0.f), 1.f); }

// bilinear sample of a 17x17 table with jax _axis semantics
__device__ __forceinline__ float bilin17(const float* __restrict__ t, float y, float x) {
  float gy = clamp01(y) * 16.f;
  float gx = clamp01(x) * 16.f;
  float fy = fminf(floorf(gy), 15.f);
  float fx = fminf(floorf(gx), 15.f);
  float wy = gy - fy, wx = gx - fx;
  int y0 = (int)fy, x0 = (int)fx;
  const float* r0 = t + y0 * 17 + x0;
  const float* r1 = r0 + 17;
  float a = r0[0], b = r0[1], c = r1[0], d = r1[1];
  return a + wx * (b - a) + wy * (c - a) + wy * wx * (a - b - c + d);
}

// ---------------- resize (jax.image.resize bilinear, antialias triangle) ----------------
// pass 1: (B,3,1080,1920) -> (B,3,1080,256) along W
__global__ void resize_w_kernel(const float* __restrict__ in, float* __restrict__ out) {
  int idx = blockIdx.x * blockDim.x + threadIdx.x;
  const int total = B_ * 3 * IMH * 256;
  if (idx >= total) return;
  int ox = idx % 256;
  int rc = idx / 256;                       // (b,c,y)
  const float s   = 256.0f / 1920.0f;
  const float inv = 1920.0f / 256.0f;       // 7.5 support radius
  float cx = (ox + 0.5f) * inv - 0.5f;
  int j0 = (int)ceilf(cx - inv); if (j0 < 0) j0 = 0;
  int j1 = (int)floorf(cx + inv); if (j1 > IMW - 1) j1 = IMW - 1;
  const float* row = in + (size_t)rc * IMW;
  float acc = 0.f, wsum = 0.f;
  for (int j = j0; j <= j1; ++j) {
    float w = 1.f - fabsf((cx - (float)j) * s);
    if (w > 0.f) { acc += w * row[j]; wsum += w; }
  }
  out[idx] = acc / wsum;
}

// pass 2: (B,3,1080,256) -> (B,3,256,256) along H
__global__ void resize_h_kernel(const float* __restrict__ in, float* __restrict__ out) {
  int idx = blockIdx.x * blockDim.x + threadIdx.x;
  const int total = B_ * 3 * 256 * 256;
  if (idx >= total) return;
  int ox = idx % 256;
  int oy = (idx / 256) % 256;
  int bc = idx / (256 * 256);
  const float s   = 256.0f / 1080.0f;
  const float inv = 1080.0f / 256.0f;       // 4.21875
  float cy = (oy + 0.5f) * inv - 0.5f;
  int j0 = (int)ceilf(cy - inv); if (j0 < 0) j0 = 0;
  int j1 = (int)floorf(cy + inv); if (j1 > IMH - 1) j1 = IMH - 1;
  const float* col = in + (size_t)bc * IMH * 256 + ox;
  float acc = 0.f, wsum = 0.f;
  for (int j = j0; j <= j1; ++j) {
    float w = 1.f - fabsf((cy - (float)j) * s);
    if (w > 0.f) { acc += w * col[(size_t)j * 256]; wsum += w; }
  }
  out[idx] = acc / wsum;
}

// ---------------- conv layer 0 (3 -> 8), too skinny for WMMA ----------------
__global__ void conv0_kernel(const float* __restrict__ in, const float* __restrict__ w,
                             const float* __restrict__ bias, float* __restrict__ out) {
  int idx = blockIdx.x * blockDim.x + threadIdx.x;
  const int total = B_ * 8 * 128 * 128;
  if (idx >= total) return;
  int ox = idx % 128;
  int oy = (idx / 128) % 128;
  int co = (idx / (128 * 128)) % 8;
  int b  = idx / (8 * 128 * 128);
  float acc = bias[co];
  for (int ci = 0; ci < 3; ++ci)
    for (int ky = 0; ky < 3; ++ky) {
      int iy = oy * 2 - 1 + ky;
      if ((unsigned)iy >= 256u) continue;
      for (int kx = 0; kx < 3; ++kx) {
        int ix = ox * 2 - 1 + kx;
        if ((unsigned)ix >= 256u) continue;
        acc += in[(((size_t)b * 3 + ci) * 256 + iy) * 256 + ix] *
               w[((co * 3 + ci) * 3 + ky) * 3 + kx];
      }
    }
  out[idx] = lrelu_(acc);
}

// ---------------- WMMA implicit-GEMM conv (layers 1..4) ----------------
// D = A(16 pixels x K) * B(K x 16 out-ch), K = Ci*9 zero-padded to Kpad (mult of 32).
// One wave (32 threads) per 16x16 output tile; no divergent path around v_wmma.
__global__ void conv_wmma_kernel(const float* __restrict__ in, const float* __restrict__ wgt,
                                 const float* __restrict__ bias, float* __restrict__ out,
                                 int Ci, int Co, int Hi, int Wi, int Ho, int Wo,
                                 int Ktot, int Kpad) {
  const int lane = threadIdx.x & 31;
  const int g    = lane >> 4;     // half-wave group
  const int lm   = lane & 15;
  const int m0   = blockIdx.x * 16;
  const int n0   = blockIdx.y * 16;
  const int b    = blockIdx.z;
  const int M    = Ho * Wo;
  v8f acc = {};
  for (int k0 = 0; k0 < Kpad; k0 += 32) {
    v16h av, bv;
#pragma unroll
    for (int h = 0; h < 16; ++h) {
      // A fragment: lane holds row m0+lm; K per ISA 16-bit A layout
      int kA = k0 + ((h < 8) ? h : h + 8) + 8 * g;
      float va = 0.f;
      int m = m0 + lm;
      if (kA < Ktot && m < M) {
        int ci = kA / 9, r = kA - ci * 9;
        int ky = r / 3, kx = r - ky * 3;
        int oy = m / Wo, ox = m - oy * Wo;
        int iy = oy * 2 - 1 + ky, ix = ox * 2 - 1 + kx;
        if ((unsigned)iy < (unsigned)Hi && (unsigned)ix < (unsigned)Wi)
          va = in[(((size_t)b * Ci + ci) * Hi + iy) * Wi + ix];
      }
      av[h] = (_Float16)va;
      // B fragment: lane holds out-channel column n0+lm
      int kB = k0 + ((h < 8) ? h : h + 8) + 8 * g;
      float vb = 0.f;
      if (kB < Ktot) vb = wgt[(size_t)(n0 + lm) * Ktot + kB];   // (Co,Ci,3,3) flat
      bv[h] = (_Float16)vb;
    }
    acc = __builtin_amdgcn_wmma_f32_16x16x32_f16(false, av, false, bv,
                                                 (short)0, acc, false, false);
  }
#pragma unroll
  for (int r = 0; r < 8; ++r) {
    int m  = m0 + r + 8 * g;        // C/D layout: VGPR r, lane group g -> row
    int co = n0 + lm;
    if (m < M) {
      float v = lrelu_(acc[r] + bias[co]);
      int oy = m / Wo;
      out[(((size_t)b * Co + co) * Ho + oy) * Wo + (m - oy * Wo)] = v;
    }
  }
}

// ---------------- instance norm (per b,c over spatial), in place ----------------
__global__ void inorm_kernel(float* __restrict__ x, const float* __restrict__ gamma,
                             const float* __restrict__ beta, int C, int HWs) {
  int bc = blockIdx.x;
  int c  = bc % C;
  float* p = x + (size_t)bc * HWs;
  __shared__ float ssum[256], ssq[256];
  float s = 0.f, q = 0.f;
  for (int i = threadIdx.x; i < HWs; i += 256) { float v = p[i]; s += v; q += v * v; }
  ssum[threadIdx.x] = s; ssq[threadIdx.x] = q;
  __syncthreads();
  for (int off = 128; off > 0; off >>= 1) {
    if (threadIdx.x < off) {
      ssum[threadIdx.x] += ssum[threadIdx.x + off];
      ssq[threadIdx.x]  += ssq[threadIdx.x + off];
    }
    __syncthreads();
  }
  float mu  = ssum[0] / (float)HWs;
  float var = ssq[0] / (float)HWs - mu * mu;
  float rs  = rsqrtf(var + 1e-5f);
  float ga = gamma[c], be = beta[c];
  for (int i = threadIdx.x; i < HWs; i += 256) p[i] = (p[i] - mu) * rs * ga + be;
}

// ---------------- 5x5/2 VALID avg pool on (B,64,8,8) -> feat (B,256) ----------------
__global__ void pool_kernel(const float* __restrict__ in, float* __restrict__ feat) {
  int idx = blockIdx.x * blockDim.x + threadIdx.x;
  if (idx >= B_ * 256) return;
  int f = idx % 256, b = idx / 256;
  int c = f / 4, r = f % 4, ph = r / 2, pw = r % 2;
  const float* p = in + ((size_t)b * 64 + c) * 64;
  float s = 0.f;
  for (int dy = 0; dy < 5; ++dy)
    for (int dx = 0; dx < 5; ++dx)
      s += p[(ph * 2 + dy) * 8 + (pw * 2 + dx)];
  feat[idx] = s * (1.f / 25.f);
}

// ---------------- generic tiny GEMV: y[b,o] = x[b,:]·W[o,:] + bias[o] ----------------
__global__ void gemv_kernel(const float* __restrict__ x, const float* __restrict__ Wm,
                            const float* __restrict__ bias, float* __restrict__ y,
                            int IN, int OUT) {
  int o = blockIdx.x * blockDim.x + threadIdx.x;
  int b = blockIdx.y;
  if (o >= OUT) return;
  const float* xb = x + (size_t)b * IN;
  const float* wr = Wm + (size_t)o * IN;
  float acc = bias[o];
  for (int i = 0; i < IN; ++i) acc += xb[i] * wr[i];
  y[(size_t)b * OUT + o] = acc;
}

// ---------------- luts[n] = U diag(sv) Vt  (n = B*9, 17x17 each) ----------------
__global__ void luts_kernel(const float* __restrict__ svd, float* __restrict__ luts) {
  int idx = blockIdx.x * blockDim.x + threadIdx.x;
  if (idx >= B_ * 9 * 289) return;
  int n = idx / 289, r = idx % 289;
  int v = r / 17, w = r % 17;
  const float* s = svd + (size_t)n * NSVD;
  float acc = 0.f;
#pragma unroll
  for (int k = 0; k < S_; ++k)
    acc += s[v * S_ + k] * s[V_ * S_ + k] * s[V_ * S_ + S_ + k * V_ + w];
  luts[idx] = acc;
}

// ---------------- fused per-pixel stage (bandwidth-bound; tables in LDS) ----------------
__global__ void pixel_kernel(const float* __restrict__ img, const float* __restrict__ luts,
                             const float* __restrict__ wb, const float* __restrict__ grids,
                             const float* __restrict__ gwb, float* __restrict__ out) {
  __shared__ float sl[9 * 289];     // 9 LUT tables
  __shared__ float sg[18 * 289];    // CPG*3 chan * 3 kind grid tables
  __shared__ float slw[9], slb[3], sgw[18], sgb[3];
  const int b = blockIdx.y;
  for (int i = threadIdx.x; i < 9 * 289; i += blockDim.x)
    sl[i] = luts[(size_t)b * 9 * 289 + i];
  for (int i = threadIdx.x; i < 18 * 289; i += blockDim.x)
    sg[i] = grids[(size_t)b * 18 * 289 + i];
  if (threadIdx.x < 9)
    slw[threadIdx.x] = wb[b * 12 + (threadIdx.x / 3) * 4 + (threadIdx.x % 3)];
  if (threadIdx.x < 18) {
    int p = threadIdx.x / 9, r = threadIdx.x % 9;
    sgw[threadIdx.x] = gwb[(b * 2 + p) * 12 + r];
  }
  if (threadIdx.x < 3) {
    slb[threadIdx.x] = wb[b * 12 + threadIdx.x * 4 + 3];
    sgb[threadIdx.x] = gwb[(b * 2 + 0) * 12 + 9 + threadIdx.x] +
                       gwb[(b * 2 + 1) * 12 + 9 + threadIdx.x];
  }
  __syncthreads();

  const size_t plane = (size_t)IMH * IMW;
  const float* imb = img + (size_t)b * 3 * plane;
  float* ob = out + (size_t)b * 3 * plane;
  const int npix = IMH * IMW;
  const int stride = gridDim.x * blockDim.x;
  for (int pix = blockIdx.x * blockDim.x + threadIdx.x; pix < npix; pix += stride) {
    __builtin_prefetch(imb + pix + stride, 0, 0);          // global_prefetch_b8
    int y = pix / IMW, x = pix - (pix / IMW) * IMW;
    float r  = clamp01(imb[pix]);
    float gg = clamp01(imb[plane + pix]);
    float bb = clamp01(imb[2 * plane + pix]);
    // coordinate pairs j: (x=r,y=g), (x=r,y=b), (x=g,y=b)
    float xc[3] = { r, r, gg };
    float yc[3] = { gg, bb, bb };
    float inten[3];
#pragma unroll
    for (int i = 0; i < 3; ++i) {
      float a = slb[i];
#pragma unroll
      for (int j = 0; j < 3; ++j)
        a += bilin17(&sl[(i * 3 + j) * 289], yc[j], xc[j]) * slw[i * 3 + j];
      inten[i] = clamp01(a);
    }
    float xg = (float)x * (1.0f / (IMW - 1));
    float yg = (float)y * (1.0f / (IMH - 1));
#pragma unroll
    for (int c = 0; c < 3; ++c) {
      float o = sgb[c];
#pragma unroll
      for (int p = 0; p < CPG_; ++p) {
        const float* t = &sg[((p * 3 + c) * 3) * 289];
        o += bilin17(t,           yg,       xg) * sgw[p * 9 + c * 3 + 0];
        o += bilin17(t + 289,     inten[c], xg) * sgw[p * 9 + c * 3 + 1];
        o += bilin17(t + 2 * 289, inten[c], yg) * sgw[p * 9 + c * 3 + 2];
      }
      ob[(size_t)c * plane + pix] = o;
    }
  }
}

// ======================= host launch =======================
extern "C" void kernel_launch(void* const* d_in, const int* in_sizes, int n_in,
                              void* d_out, int out_size, void* d_ws, size_t ws_size,
                              hipStream_t stream) {
  (void)in_sizes; (void)n_in; (void)out_size; (void)ws_size;
  const float* img = (const float*)d_in[0];
  const float* cw[5] = { (const float*)d_in[1], (const float*)d_in[2], (const float*)d_in[3],
                         (const float*)d_in[4], (const float*)d_in[5] };
  const float* cb[5] = { (const float*)d_in[6], (const float*)d_in[7], (const float*)d_in[8],
                         (const float*)d_in[9], (const float*)d_in[10] };
  const float* ing[4] = { (const float*)d_in[11], (const float*)d_in[12],
                          (const float*)d_in[13], (const float*)d_in[14] };
  const float* inb[4] = { (const float*)d_in[15], (const float*)d_in[16],
                          (const float*)d_in[17], (const float*)d_in[18] };
  const float* lut_Wg = (const float*)d_in[19];  const float* lut_bg = (const float*)d_in[20];
  const float* lut_Wb = (const float*)d_in[21];  const float* lut_bb = (const float*)d_in[22];
  const float* lwb_Wg = (const float*)d_in[23];  const float* lwb_bg = (const float*)d_in[24];
  const float* lwb_Wb = (const float*)d_in[25];  const float* lwb_bb = (const float*)d_in[26];
  const float* grid_Wg = (const float*)d_in[27]; const float* grid_bg = (const float*)d_in[28];
  const float* grid_Wb = (const float*)d_in[29]; const float* grid_bb = (const float*)d_in[30];
  const float* gwb_Wg = (const float*)d_in[31];  const float* gwb_bg = (const float*)d_in[32];
  const float* gwb_Wb = (const float*)d_in[33];  const float* gwb_bb = (const float*)d_in[34];
  float* out = (float*)d_out;
  float* ws  = (float*)d_ws;

  // workspace layout (floats); conv ping-pong reuses the resize temp region
  const size_t O_TMP = 0;                              // B*3*1080*256 = 1,658,880
  const size_t O_RES = O_TMP + (size_t)B_ * 3 * IMH * 256;   // B*3*256*256 = 393,216
  const size_t O_FEAT = O_RES + (size_t)B_ * 3 * 256 * 256;  // 512
  const size_t O_W1 = O_FEAT + B_ * 256;    // B*24
  const size_t O_W2 = O_W1 + B_ * 24;       // B*8
  const size_t O_W3 = O_W2 + B_ * 8;        // B*24
  const size_t O_W4 = O_W3 + B_ * 24;       // B*8
  const size_t O_SVD = O_W4 + B_ * 8;       // B*2520
  const size_t O_LUT = O_SVD + (size_t)B_ * 9 * NSVD;  // B*9*289
  const size_t O_WB  = O_LUT + (size_t)B_ * 9 * 289;   // B*12
  const size_t O_GRD = O_WB + B_ * 12;                 // B*CPG*9*289
  const size_t O_GWB = O_GRD + (size_t)B_ * CPG_ * 9 * 289;  // B*CPG*12

  float* tmpW  = ws + O_TMP;
  float* resz  = ws + O_RES;
  float* actA  = ws + O_TMP;                 // reused after resize complete
  float* actB  = ws + O_TMP + (size_t)B_ * 8 * 128 * 128;
  float* feat  = ws + O_FEAT;
  float* svd   = ws + O_SVD;
  float* lutsB = ws + O_LUT;
  float* wbB   = ws + O_WB;
  float* grdB  = ws + O_GRD;
  float* gwbB  = ws + O_GWB;

  // 1) antialiased resize to 256x256
  {
    int n = B_ * 3 * IMH * 256;
    resize_w_kernel<<<(n + 255) / 256, 256, 0, stream>>>(img, tmpW);
    n = B_ * 3 * 256 * 256;
    resize_h_kernel<<<(n + 255) / 256, 256, 0, stream>>>(tmpW, resz);
  }
  // 2) backbone: conv0 (VALU) + WMMA convs 1..4, instance norm between
  {
    int n = B_ * 8 * 128 * 128;
    conv0_kernel<<<(n + 255) / 256, 256, 0, stream>>>(resz, cw[0], cb[0], actA);
    inorm_kernel<<<B_ * 8, 256, 0, stream>>>(actA, ing[0], inb[0], 8, 128 * 128);

    conv_wmma_kernel<<<dim3(256, 1, B_), 32, 0, stream>>>(actA, cw[1], cb[1], actB,
        8, 16, 128, 128, 64, 64, 72, 96);
    inorm_kernel<<<B_ * 16, 256, 0, stream>>>(actB, ing[1], inb[1], 16, 64 * 64);

    conv_wmma_kernel<<<dim3(64, 2, B_), 32, 0, stream>>>(actB, cw[2], cb[2], actA,
        16, 32, 64, 64, 32, 32, 144, 160);
    inorm_kernel<<<B_ * 32, 256, 0, stream>>>(actA, ing[2], inb[2], 32, 32 * 32);

    conv_wmma_kernel<<<dim3(16, 4, B_), 32, 0, stream>>>(actA, cw[3], cb[3], actB,
        32, 64, 32, 32, 16, 16, 288, 288);
    inorm_kernel<<<B_ * 64, 256, 0, stream>>>(actB, ing[3], inb[3], 64, 16 * 16);

    conv_wmma_kernel<<<dim3(4, 4, B_), 32, 0, stream>>>(actB, cw[4], cb[4], actA,
        64, 64, 16, 16, 8, 8, 576, 576);

    pool_kernel<<<(B_ * 256 + 255) / 256, 256, 0, stream>>>(actA, feat);
  }
  // 3) heads (tiny GEMVs)
  {
    gemv_kernel<<<dim3(1, B_), 256, 0, stream>>>(feat, lut_Wg, lut_bg, ws + O_W1, 256, 24);
    gemv_kernel<<<dim3((9 * NSVD + 255) / 256, B_), 256, 0, stream>>>(ws + O_W1, lut_Wb, lut_bb, svd, 24, 9 * NSVD);
    gemv_kernel<<<dim3(1, B_), 256, 0, stream>>>(feat, lwb_Wg, lwb_bg, ws + O_W2, 256, 8);
    gemv_kernel<<<dim3(1, B_), 256, 0, stream>>>(ws + O_W2, lwb_Wb, lwb_bb, wbB, 8, 12);
    gemv_kernel<<<dim3(1, B_), 256, 0, stream>>>(feat, grid_Wg, grid_bg, ws + O_W3, 256, 24);
    gemv_kernel<<<dim3((CPG_ * 9 * 289 + 255) / 256, B_), 256, 0, stream>>>(ws + O_W3, grid_Wb, grid_bb, grdB, 24, CPG_ * 9 * 289);
    gemv_kernel<<<dim3(1, B_), 256, 0, stream>>>(feat, gwb_Wg, gwb_bg, ws + O_W4, 256, 8);
    gemv_kernel<<<dim3(1, B_), 256, 0, stream>>>(ws + O_W4, gwb_Wb, gwb_bb, gwbB, 8, CPG_ * 12);
    int n = B_ * 9 * 289;
    luts_kernel<<<(n + 255) / 256, 256, 0, stream>>>(svd, lutsB);
  }
  // 4) fused per-pixel stage (HBM-bound streaming, tables in LDS)
  pixel_kernel<<<dim3(2048, B_), 256, 0, stream>>>(img, lutsB, wbB, grdB, gwbB, out);
}
